// AttentionPairBias_515396076359
// MI455X (gfx1250) — matrix operand order repaired and checked by reading, not compiled, not run on hardware
//
#include <hip/hip_runtime.h>
#include <hip/hip_bf16.h>

typedef __attribute__((ext_vector_type(16))) _Float16 v16h;
typedef __attribute__((ext_vector_type(8)))  _Float16 v8h;
typedef __attribute__((ext_vector_type(4)))  _Float16 v4h;
typedef __attribute__((ext_vector_type(8)))  float    v8f;
typedef __attribute__((ext_vector_type(4)))  float    v4f;

#define SS   1024   // sequence length
#define DD   768    // d_model
#define HH   16     // heads
#define HD   48     // head dim
#define HDP  64     // padded head dim
#define DZ   128    // pair channel dim

// ---------------------------------------------------------------------------
// WMMA fragment loaders (layouts per CDNA5 ISA, 16-bit A/B, wave32)
// ---------------------------------------------------------------------------
__device__ __forceinline__ v16h ld_pair(const _Float16* p0, const _Float16* p1) {
  v8h a = *(const v8h*)p0;
  v8h b = *(const v8h*)p1;
  v16h r;
#pragma unroll
  for (int i = 0; i < 8; ++i) { r[i] = a[i]; r[i + 8] = b[i]; }
  return r;
}

// A fragment: 16(M) x 32(K) tile of row-major matrix [*, ldk], K offset kk.
// lane<16: M=lane, halves = K kk+0..7 , kk+16..23
// lane>=16: M=lane-16, halves = K kk+8..15, kk+24..31
__device__ __forceinline__ v16h ld_a_frag(const _Float16* base, int ldk, int kk, int lane) {
  const int hlf = lane >> 4, mr = lane & 15;
  const _Float16* p = base + (size_t)mr * ldk + kk + hlf * 8;
  return ld_pair(p, p + 16);
}

// B fragment: 32(K) x 16(N) with B[k][n] = Wrow[n][k] (row-major [*, ldk]).
// lane<16: N=lane, halves = K kk+0..7 , kk+16..23; lane>=16: N=lane-16, halves = K kk+8..15, kk+24..31
__device__ __forceinline__ v16h ld_b_frag(const _Float16* base, int ldk, int kk, int lane) {
  const int hlf = lane >> 4, nr = lane & 15;
  const _Float16* p = base + (size_t)nr * ldk + kk + hlf * 8;
  return ld_pair(p, p + 16);
}

// ---------------------------------------------------------------------------
// Utility kernels
// ---------------------------------------------------------------------------
__global__ void cvt_f32_to_f16(const float* __restrict__ x, _Float16* __restrict__ y, int n) {
  int i = (blockIdx.x * blockDim.x + threadIdx.x) * 4;
  if (i < n) {
    v4f v = *(const v4f*)(x + i);
    v4h o; o[0] = (_Float16)v.x; o[1] = (_Float16)v.y; o[2] = (_Float16)v.z; o[3] = (_Float16)v.w;
    *(v4h*)(y + i) = o;
  }
}

__global__ void zero_f16(_Float16* __restrict__ y, int n) {
  int i = (blockIdx.x * blockDim.x + threadIdx.x) * 8;
  if (i < n) {
    v8h zv = {};
    *(v8h*)(y + i) = zv;
  }
}

// ---------------------------------------------------------------------------
// Fused QKV+G projection. grid=(64, 6, 4) block=256 (8 waves).
// which: 0=q (+bias, *1/sqrt(HD), -> q_h[h][s][HDP]), 1=k, 2=v (-> vT[h][d][s]),
//        3=g (-> g_f[s][n] fp32).
// ---------------------------------------------------------------------------
__global__ __launch_bounds__(256) void proj_kernel(
    const _Float16* __restrict__ s_h,
    const _Float16* __restrict__ Wq_h, const _Float16* __restrict__ Wk_h,
    const _Float16* __restrict__ Wv_h, const _Float16* __restrict__ Wg_h,
    const float* __restrict__ bq,
    _Float16* __restrict__ q_h, _Float16* __restrict__ k_h,
    _Float16* __restrict__ vT_h, float* __restrict__ g_f) {
  const int lane = threadIdx.x & 31;
  const int wave = threadIdx.x >> 5;
  const int m0 = blockIdx.x * 16;
  const int n0 = (blockIdx.y * 8 + wave) * 16;
  const int which = blockIdx.z;
  const _Float16* W = (which == 0) ? Wq_h : (which == 1) ? Wk_h : (which == 2) ? Wv_h : Wg_h;

  const _Float16* abase = s_h + (size_t)m0 * DD;
  const _Float16* bbase = W + (size_t)n0 * DD;
  v8f c = {};
#pragma unroll 4
  for (int kk = 0; kk < DD; kk += 32) {
    v16h a = ld_a_frag(abase, DD, kk, lane);
    v16h b = ld_b_frag(bbase, DD, kk, lane);
    c = __builtin_amdgcn_wmma_f32_16x16x32_f16(false, a, false, b, (short)0, c, false, false);
  }

  const int hlf = lane >> 4, nn = lane & 15;
  const int n = n0 + nn;            // global output column; 16-tiles never cross head bounds
  const int h = n / HD, d = n % HD;
  if (which == 0) {
    const float scale = 0.14433756729740643f;  // HD^-0.5
    const float bb = bq[n];
#pragma unroll
    for (int r = 0; r < 8; ++r) {
      int srow = m0 + hlf * 8 + r;
      q_h[((size_t)h * SS + srow) * HDP + d] = (_Float16)((c[r] + bb) * scale);
    }
  } else if (which == 1) {
#pragma unroll
    for (int r = 0; r < 8; ++r) {
      int srow = m0 + hlf * 8 + r;
      k_h[((size_t)h * SS + srow) * HDP + d] = (_Float16)c[r];
    }
  } else if (which == 2) {
    v8h pk;
#pragma unroll
    for (int r = 0; r < 8; ++r) pk[r] = (_Float16)c[r];
    // rows of D (consecutive r) are consecutive s -> one 16B store per lane
    *(v8h*)(vT_h + ((size_t)h * HDP + d) * SS + m0 + hlf * 8) = pk;
  } else {
#pragma unroll
    for (int r = 0; r < 8; ++r) {
      int srow = m0 + hlf * 8 + r;
      g_f[(size_t)srow * DD + n] = c[r];
    }
  }
}

// ---------------------------------------------------------------------------
// Pair bias: LayerNorm(z[s][t][:]) . Wz[h][:] -> bias[h][s][t]  (fp32)
// One wave per 8 consecutive t of one s. grid=16384 blocks * 8 waves.
// Bandwidth-bound on z (512 MB, streamed once): non-temporal loads keep the
// reused intermediates (bias/a/q/k/vT) resident in the 192MB L2.
// ---------------------------------------------------------------------------
__global__ __launch_bounds__(256) void pairbias_kernel(
    const float* __restrict__ z, const float* __restrict__ ln_w,
    const float* __restrict__ ln_b, const float* __restrict__ Wz,
    float* __restrict__ bias) {
  const int lane = threadIdx.x & 31;
  const int wave = threadIdx.x >> 5;
  const int widx = blockIdx.x * 8 + wave;   // 0 .. 1024*128-1
  const int si = widx >> 7;
  const int t0 = (widx & 127) * 8;

  v4f lw = *(const v4f*)(ln_w + lane * 4);
  v4f lb = *(const v4f*)(ln_b + lane * 4);
  v4f wzv[16];
#pragma unroll
  for (int h = 0; h < 16; ++h) wzv[h] = *(const v4f*)(Wz + h * DZ + lane * 4);

  const float* zbase = z + ((size_t)si * SS + t0) * DZ + lane * 4;
  for (int it = 0; it < 8; ++it) {
    // prefetch next row (global_prefetch_b8), stream current row non-temporally
    if (it < 7) __builtin_prefetch(zbase + (it + 1) * DZ, 0, 1);
    v4f zv = __builtin_nontemporal_load((const v4f*)(zbase + it * DZ));
    float s1 = zv.x + zv.y + zv.z + zv.w;
    float s2 = zv.x * zv.x + zv.y * zv.y + zv.z * zv.z + zv.w * zv.w;
#pragma unroll
    for (int m = 16; m >= 1; m >>= 1) {
      s1 += __shfl_xor(s1, m, 32);
      s2 += __shfl_xor(s2, m, 32);
    }
    const float mean = s1 * (1.0f / DZ);
    const float var = s2 * (1.0f / DZ) - mean * mean;
    const float rstd = rsqrtf(var + 1e-5f);
    v4f zn;
    zn.x = (zv.x - mean) * rstd * lw.x + lb.x;
    zn.y = (zv.y - mean) * rstd * lw.y + lb.y;
    zn.z = (zv.z - mean) * rstd * lw.z + lb.z;
    zn.w = (zv.w - mean) * rstd * lw.w + lb.w;

    float p[16];
#pragma unroll
    for (int h = 0; h < 16; ++h)
      p[h] = zn.x * wzv[h].x + zn.y * wzv[h].y + zn.z * wzv[h].z + zn.w * wzv[h].w;
#pragma unroll
    for (int m = 16; m >= 1; m >>= 1) {
#pragma unroll
      for (int h = 0; h < 16; ++h) p[h] += __shfl_xor(p[h], m, 32);
    }
    float res = p[0];
#pragma unroll
    for (int h = 1; h < 16; ++h) res = (lane == h) ? p[h] : res;
    if (lane < 16) bias[((size_t)lane * SS + si) * SS + (t0 + it)] = res;
  }
}

// ---------------------------------------------------------------------------
// scores = q k^T (scale folded into q) + bias; row softmax; a -> fp16.
// grid=(64 s-tiles, 16 heads), block=256 (8 waves), 64KB LDS row buffer.
// q A-fragments hoisted: loaded once, reused across all 8 t-tiles.
// ---------------------------------------------------------------------------
__global__ __launch_bounds__(256) void attn_scores_kernel(
    const _Float16* __restrict__ q_h, const _Float16* __restrict__ k_h,
    const float* __restrict__ bias, _Float16* __restrict__ a_h) {
  __shared__ float sc[16 * SS];  // 64 KB
  const int lane = threadIdx.x & 31;
  const int wave = threadIdx.x >> 5;
  const int s0 = blockIdx.x * 16;
  const int h = blockIdx.y;
  const int hlf = lane >> 4, nn = lane & 15;
  const _Float16* qb = q_h + ((size_t)h * SS + s0) * HDP;

  const v16h qa0 = ld_a_frag(qb, HDP, 0, lane);
  const v16h qa1 = ld_a_frag(qb, HDP, 32, lane);

  for (int i = 0; i < 8; ++i) {
    const int t0 = (wave * 8 + i) * 16;
    const _Float16* kb = k_h + ((size_t)h * SS + t0) * HDP;
    v8f c = {};
    v16h b0 = ld_b_frag(kb, HDP, 0, lane);
    v16h b1 = ld_b_frag(kb, HDP, 32, lane);
    c = __builtin_amdgcn_wmma_f32_16x16x32_f16(false, qa0, false, b0, (short)0, c, false, false);
    c = __builtin_amdgcn_wmma_f32_16x16x32_f16(false, qa1, false, b1, (short)0, c, false, false);
#pragma unroll
    for (int r = 0; r < 8; ++r) {
      const int m = hlf * 8 + r;
      const float v = c[r] + bias[((size_t)h * SS + (s0 + m)) * SS + t0 + nn];
      sc[m * SS + t0 + nn] = v;
    }
  }
  __syncthreads();

  for (int rr = 0; rr < 2; ++rr) {
    const int row = wave * 2 + rr;
    float mx = -1e30f;
#pragma unroll 4
    for (int i = lane; i < SS; i += 32) mx = fmaxf(mx, sc[row * SS + i]);
#pragma unroll
    for (int m = 16; m >= 1; m >>= 1) mx = fmaxf(mx, __shfl_xor(mx, m, 32));
    float sum = 0.f;
#pragma unroll 4
    for (int i = lane; i < SS; i += 32) {
      const float e = __expf(sc[row * SS + i] - mx);
      sc[row * SS + i] = e;
      sum += e;
    }
#pragma unroll
    for (int m = 16; m >= 1; m >>= 1) sum += __shfl_xor(sum, m, 32);
    const float inv = 1.0f / sum;
    _Float16* arow = a_h + ((size_t)h * SS + s0 + row) * SS;
#pragma unroll 4
    for (int i = lane; i < SS; i += 32) arow[i] = (_Float16)(sc[row * SS + i] * inv);
  }
}

// ---------------------------------------------------------------------------
// o = a @ v (K=1024), fused sigmoid(g) gate; write og fp16 [s][DD].
// One wave computes the full 16(s) x 48(d) block: 3 accumulators share each
// A fragment (3 WMMA per A load; a-matrix read exactly once).
// grid=(8, 16), block=256.
// ---------------------------------------------------------------------------
__global__ __launch_bounds__(256) void attn_o_kernel(
    const _Float16* __restrict__ a_h, const _Float16* __restrict__ vT_h,
    const float* __restrict__ g_f, _Float16* __restrict__ og_h) {
  const int lane = threadIdx.x & 31;
  const int wave = threadIdx.x >> 5;
  const int s0 = (blockIdx.x * 8 + wave) * 16;
  const int h = blockIdx.y;
  const _Float16* ab = a_h + ((size_t)h * SS + s0) * SS;
  const _Float16* b0b = vT_h + ((size_t)h * HDP + 0) * SS;
  const _Float16* b1b = vT_h + ((size_t)h * HDP + 16) * SS;
  const _Float16* b2b = vT_h + ((size_t)h * HDP + 32) * SS;
  v8f c0 = {}, c1 = {}, c2 = {};
#pragma unroll 2
  for (int kk = 0; kk < SS; kk += 32) {
    v16h a = ld_a_frag(ab, SS, kk, lane);
    v16h b0 = ld_b_frag(b0b, SS, kk, lane);
    v16h b1 = ld_b_frag(b1b, SS, kk, lane);
    v16h b2 = ld_b_frag(b2b, SS, kk, lane);
    c0 = __builtin_amdgcn_wmma_f32_16x16x32_f16(false, a, false, b0, (short)0, c0, false, false);
    c1 = __builtin_amdgcn_wmma_f32_16x16x32_f16(false, a, false, b1, (short)0, c1, false, false);
    c2 = __builtin_amdgcn_wmma_f32_16x16x32_f16(false, a, false, b2, (short)0, c2, false, false);
  }
  const int hlf = lane >> 4, nn = lane & 15;
#pragma unroll
  for (int dt = 0; dt < 3; ++dt) {
    const v8f c = (dt == 0) ? c0 : (dt == 1) ? c1 : c2;
    const int n = h * HD + dt * 16 + nn;
#pragma unroll
    for (int r = 0; r < 8; ++r) {
      const int srow = s0 + hlf * 8 + r;
      const float gv = g_f[(size_t)srow * DD + n];
      const float sig = 1.0f / (1.0f + __expf(-gv));
      og_h[(size_t)srow * DD + n] = (_Float16)(c[r] * sig);
    }
  }
}

// ---------------------------------------------------------------------------
// out = og @ Wo^T  (fp32 output). grid=(64, 6), block=256.
// ---------------------------------------------------------------------------
__global__ __launch_bounds__(256) void out_proj_kernel(
    const _Float16* __restrict__ og_h, const _Float16* __restrict__ Wo_h,
    float* __restrict__ out) {
  const int lane = threadIdx.x & 31;
  const int wave = threadIdx.x >> 5;
  const int m0 = blockIdx.x * 16;
  const int n0 = (blockIdx.y * 8 + wave) * 16;
  const _Float16* abase = og_h + (size_t)m0 * DD;
  const _Float16* bbase = Wo_h + (size_t)n0 * DD;
  v8f c = {};
#pragma unroll 4
  for (int kk = 0; kk < DD; kk += 32) {
    v16h a = ld_a_frag(abase, DD, kk, lane);
    v16h b = ld_b_frag(bbase, DD, kk, lane);
    c = __builtin_amdgcn_wmma_f32_16x16x32_f16(false, a, false, b, (short)0, c, false, false);
  }
  const int hlf = lane >> 4, nn = lane & 15;
#pragma unroll
  for (int r = 0; r < 8; ++r) {
    const int srow = m0 + hlf * 8 + r;
    out[(size_t)srow * DD + n0 + nn] = c[r];
  }
}

// ---------------------------------------------------------------------------
extern "C" void kernel_launch(void* const* d_in, const int* in_sizes, int n_in,
                              void* d_out, int out_size, void* d_ws, size_t ws_size,
                              hipStream_t stream) {
  const float* s    = (const float*)d_in[0];
  const float* z    = (const float*)d_in[1];
  const float* Wq   = (const float*)d_in[2];
  const float* bq   = (const float*)d_in[3];
  const float* Wk   = (const float*)d_in[4];
  const float* Wv   = (const float*)d_in[5];
  const float* Wg   = (const float*)d_in[6];
  const float* ln_w = (const float*)d_in[7];
  const float* ln_b = (const float*)d_in[8];
  const float* Wz   = (const float*)d_in[9];
  const float* Wo   = (const float*)d_in[10];
  float* out = (float*)d_out;

  char* ws = (char*)d_ws;
  size_t off = 0;
  auto take = [&](size_t bytes) -> void* {
    void* p = ws + off;
    off += (bytes + 255) & ~(size_t)255;
    return p;
  };
  _Float16* s_h   = (_Float16*)take((size_t)SS * DD * 2);
  _Float16* Wq_h  = (_Float16*)take((size_t)DD * DD * 2);
  _Float16* Wk_h  = (_Float16*)take((size_t)DD * DD * 2);
  _Float16* Wv_h  = (_Float16*)take((size_t)DD * DD * 2);
  _Float16* Wg_h  = (_Float16*)take((size_t)DD * DD * 2);
  _Float16* Wo_h  = (_Float16*)take((size_t)DD * DD * 2);
  _Float16* q_h   = (_Float16*)take((size_t)HH * SS * HDP * 2);
  _Float16* k_h   = (_Float16*)take((size_t)HH * SS * HDP * 2);
  _Float16* vT_h  = (_Float16*)take((size_t)HH * HDP * SS * 2);
  float*    g_f   = (float*)take((size_t)SS * DD * 4);
  float*    biasf = (float*)take((size_t)HH * SS * SS * 4);
  _Float16* a_h   = (_Float16*)take((size_t)HH * SS * SS * 2);
  _Float16* og_h  = (_Float16*)take((size_t)SS * DD * 2);

  // 1) fp32 -> fp16 conversions
  cvt_f32_to_f16<<<(SS * DD / 4 + 255) / 256, 256, 0, stream>>>(s, s_h, SS * DD);
  cvt_f32_to_f16<<<(DD * DD / 4 + 255) / 256, 256, 0, stream>>>(Wq, Wq_h, DD * DD);
  cvt_f32_to_f16<<<(DD * DD / 4 + 255) / 256, 256, 0, stream>>>(Wk, Wk_h, DD * DD);
  cvt_f32_to_f16<<<(DD * DD / 4 + 255) / 256, 256, 0, stream>>>(Wv, Wv_h, DD * DD);
  cvt_f32_to_f16<<<(DD * DD / 4 + 255) / 256, 256, 0, stream>>>(Wg, Wg_h, DD * DD);
  cvt_f32_to_f16<<<(DD * DD / 4 + 255) / 256, 256, 0, stream>>>(Wo, Wo_h, DD * DD);

  // 2) zero padded q/k/vT (pad cols 48..63 must be 0 for the padded-K WMMA)
  zero_f16<<<(HH * SS * HDP / 8 + 255) / 256, 256, 0, stream>>>(q_h, HH * SS * HDP);
  zero_f16<<<(HH * SS * HDP / 8 + 255) / 256, 256, 0, stream>>>(k_h, HH * SS * HDP);
  zero_f16<<<(HH * SS * HDP / 8 + 255) / 256, 256, 0, stream>>>(vT_h, HH * HDP * SS);

  // 3) fused QKV + gate projections (WMMA)
  proj_kernel<<<dim3(SS / 16, DD / 128, 4), 256, 0, stream>>>(
      s_h, Wq_h, Wk_h, Wv_h, Wg_h, bq, q_h, k_h, vT_h, g_f);

  // 4) pair bias (bandwidth-bound: 512 MB of z, non-temporal stream)
  pairbias_kernel<<<SS * (SS / 8) / 8, 256, 0, stream>>>(z, ln_w, ln_b, Wz, biasf);

  // 5) scores + bias + softmax -> a (WMMA + LDS softmax)
  attn_scores_kernel<<<dim3(SS / 16, HH), 256, 0, stream>>>(q_h, k_h, biasf, a_h);

  // 6) o = a @ v, fused sigmoid gate (WMMA, 3 N-tiles per wave)
  attn_o_kernel<<<dim3(SS / 16 / 8, HH), 256, 0, stream>>>(a_h, vT_h, g_f, og_h);

  // 7) out = og @ Wo^T (WMMA, fp32 out)
  out_proj_kernel<<<dim3(SS / 16, DD / 128), 256, 0, stream>>>(og_h, Wo_h, out);
}
